// SelfAttention_5763846111404
// MI455X (gfx1250) — matrix operand order repaired and checked by reading, not compiled, not run on hardware
//
#include <hip/hip_runtime.h>
#include <hip/hip_bf16.h>

// ---------------------------------------------------------------------------
// Self-attention forward for MI455X (gfx1250), bf16 WMMA path.
// A-side buffers are stored in true K order (lane half h reads K 16h..16h+15 of each
// 32-element K chunk) so every WMMA fragment is ONE aligned 32-byte load.
// B=2, S=2048, E=1024, H=16, D=64.
// ---------------------------------------------------------------------------

typedef __attribute__((ext_vector_type(16))) __bf16 v16bf;
typedef __attribute__((ext_vector_type(8)))  float  v8f;

#define WMMA_BF16(a, b, c) \
  __builtin_amdgcn_wmma_f32_16x16x32_bf16(false, (a), false, (b), (short)0, (c), false, false)

// ---- constants ----
#define BATCH 2
#define SEQ   2048
#define EMB   1024
#define HEADS 16
#define HDIM  64
#define ROWS  (BATCH * SEQ)      // 4096
#define N3E   (3 * EMB)          // 3072
#define KCHUNKS (EMB / 32)       // 32

// Involutive swizzle inside a 32-element chunk: swap groups [8..15] <-> [16..23].
// Packed A rows are not swizzled: positions 0..15 hold true K 0..15 (= lanes 0-15
// fragment data) and 16..31 hold true K 16..31 (= lanes 16-31 data).
__device__ __forceinline__ int swz32(int c) {
  const int g = (c >> 3) & 3;
  return c + ((g == 1) ? 8 : (g == 2) ? -8 : 0);
}

// ---------------------------------------------------------------------------
// Fragment loaders: one 32-byte load each (splits into 2x global_load_b128
// writing one contiguous 8-VGPR group).
// A: from K-contiguous (row-major) storage. B: from K-contiguous (transposed) storage.
// ---------------------------------------------------------------------------
__device__ __forceinline__ v16bf load_frag_a(const __bf16* base, int ldr, int lane) {
  return *(const v16bf*)(base + (lane & 15) * ldr + ((lane < 16) ? 0 : 16));
}

__device__ __forceinline__ v16bf load_frag_b(const __bf16* baseT, int ldr, int lane) {
  return *(const v16bf*)(baseT + (lane & 15) * ldr + ((lane < 16) ? 0 : 16));
}

// ---------------------------------------------------------------------------
// Kernel 1: convert x -> bf16 (true K order), transpose+convert W_in, W_out.
// ---------------------------------------------------------------------------
__global__ void attn_pack_kernel(const float* __restrict__ x,
                                 const float* __restrict__ Win,
                                 const float* __restrict__ Wout,
                                 __bf16* __restrict__ xb,
                                 __bf16* __restrict__ winT,
                                 __bf16* __restrict__ woutT) {
  int idx = blockIdx.x * 256 + threadIdx.x;
  const int NX = ROWS * EMB;
  const int NWIN = EMB * N3E;
  const int NWOUT = EMB * EMB;
  if (idx < NX) {                    // xb[row][k] = x[row][k]
    const int k = idx & (EMB - 1);
    const int dst = idx;
    xb[dst] = (__bf16)x[idx];
    return;
  }
  idx -= NX;
  if (idx < NWIN) {                  // winT[j][k] = Win[k][j]  (true-K order)
    int j = idx >> 10, k = idx & 1023;
    winT[idx] = (__bf16)Win[k * N3E + j];
    return;
  }
  idx -= NWIN;
  if (idx < NWOUT) {                 // woutT[j][k] = Wout[k][j]
    int j = idx >> 10, k = idx & 1023;
    woutT[idx] = (__bf16)Wout[k * EMB + j];
  }
}

// ---------------------------------------------------------------------------
// Shared ping-pong GEMM core: 32x64 output tile per wave, K = EMB.
// ---------------------------------------------------------------------------
__device__ __forceinline__ void gemm_core_32x64(const __bf16* __restrict__ arow0,
                                                const __bf16* __restrict__ arow1,
                                                const __bf16* __restrict__ bbase,
                                                int lane, v8f acc0[4], v8f acc1[4]) {
  v16bf A0[2], A1[2], Bf[2][4];
  A0[0] = load_frag_a(arow0, EMB, lane);
  A1[0] = load_frag_a(arow1, EMB, lane);
#pragma unroll
  for (int t = 0; t < 4; ++t) Bf[0][t] = load_frag_b(bbase + t * 16 * EMB, EMB, lane);

#pragma unroll 1
  for (int c = 0; c < KCHUNKS - 2; c += 2) {
    const int k1 = (c + 1) * 32;
    A0[1] = load_frag_a(arow0 + k1, EMB, lane);
    A1[1] = load_frag_a(arow1 + k1, EMB, lane);
#pragma unroll
    for (int t = 0; t < 4; ++t) Bf[1][t] = load_frag_b(bbase + t * 16 * EMB + k1, EMB, lane);
    __builtin_prefetch((const void*)(arow0 + k1 + 96));   // global_prefetch_b8
    __builtin_prefetch((const void*)(arow1 + k1 + 96));
#pragma unroll
    for (int t = 0; t < 4; ++t) {
      acc0[t] = WMMA_BF16(A0[0], Bf[0][t], acc0[t]);
      acc1[t] = WMMA_BF16(A1[0], Bf[0][t], acc1[t]);
    }
    const int k2 = (c + 2) * 32;
    A0[0] = load_frag_a(arow0 + k2, EMB, lane);
    A1[0] = load_frag_a(arow1 + k2, EMB, lane);
#pragma unroll
    for (int t = 0; t < 4; ++t) Bf[0][t] = load_frag_b(bbase + t * 16 * EMB + k2, EMB, lane);
#pragma unroll
    for (int t = 0; t < 4; ++t) {
      acc0[t] = WMMA_BF16(A0[1], Bf[1][t], acc0[t]);
      acc1[t] = WMMA_BF16(A1[1], Bf[1][t], acc1[t]);
    }
  }
  {
    const int k1 = (KCHUNKS - 1) * 32;
    A0[1] = load_frag_a(arow0 + k1, EMB, lane);
    A1[1] = load_frag_a(arow1 + k1, EMB, lane);
#pragma unroll
    for (int t = 0; t < 4; ++t) Bf[1][t] = load_frag_b(bbase + t * 16 * EMB + k1, EMB, lane);
#pragma unroll
    for (int t = 0; t < 4; ++t) {
      acc0[t] = WMMA_BF16(A0[0], Bf[0][t], acc0[t]);
      acc1[t] = WMMA_BF16(A1[0], Bf[0][t], acc1[t]);
    }
#pragma unroll
    for (int t = 0; t < 4; ++t) {
      acc0[t] = WMMA_BF16(A0[1], Bf[1][t], acc0[t]);
      acc1[t] = WMMA_BF16(A1[1], Bf[1][t], acc1[t]);
    }
  }
}

// ---------------------------------------------------------------------------
// QKV scatter store of one 16x64 C strip (4 tiles).
// Q (A-operand of QK^T), K and V^T are all stored in true d order.
// ---------------------------------------------------------------------------
__device__ __forceinline__ void qkv_store(const v8f* acc, int i0, int j0, int lane,
                                          const float* __restrict__ b_in,
                                          __bf16* __restrict__ qout,
                                          __bf16* __restrict__ kout,
                                          __bf16* __restrict__ vTout) {
  const int n = lane & 15;
  const int mrow0 = (lane < 16) ? 0 : 8;
  const int which = j0 >> 10;                        // 0=q 1=k 2=v
#pragma unroll
  for (int t = 0; t < 4; ++t) {
    const int j = j0 + t * 16 + n;
    const int e = j & (EMB - 1);
    const int h = e >> 6, d = e & 63;
    const int dswz = (d & ~31) | swz32(d & 31);
    const float bv = b_in[j];
#pragma unroll
    for (int r = 0; r < 8; ++r) {
      const int i = i0 + mrow0 + r;
      const int bb = i >> 11;                        // i / SEQ
      const int s = i & (SEQ - 1);
      const float val = acc[t][r] + bv;
      if (which == 0) {
        qout[((bb * HEADS + h) * SEQ + s) * HDIM + d] = (__bf16)(val * 0.125f);
      } else if (which == 1) {
        kout[((bb * HEADS + h) * SEQ + s) * HDIM + d] = (__bf16)val;
      } else {
        vTout[((bb * HEADS + h) * HDIM + d) * SEQ + s] = (__bf16)val;
      }
    }
  }
}

// ---------------------------------------------------------------------------
// Kernel 2: QKV projection, 32x64 tile per wave.
// ---------------------------------------------------------------------------
__global__ __launch_bounds__(128)
void attn_qkv_gemm_kernel(const __bf16* __restrict__ xb,
                          const __bf16* __restrict__ winT,
                          const float* __restrict__ b_in,
                          __bf16* __restrict__ qout,
                          __bf16* __restrict__ kout,
                          __bf16* __restrict__ vTout) {
  const int lane = threadIdx.x & 31;
  const int wid  = threadIdx.x >> 5;
  const int i0 = blockIdx.x * 32;
  const int j0 = (blockIdx.y * 4 + wid) * 64;

  v8f acc0[4] = {}, acc1[4] = {};
  gemm_core_32x64(xb + i0 * EMB, xb + (i0 + 16) * EMB, winT + j0 * EMB,
                  lane, acc0, acc1);

  qkv_store(acc0, i0,      j0, lane, b_in, qout, kout, vTout);
  qkv_store(acc1, i0 + 16, j0, lane, b_in, qout, kout, vTout);
}

// ---------------------------------------------------------------------------
// Kernel 3: flash attention, one wave per (bh, 16-row Q tile).
// ---------------------------------------------------------------------------
__global__ __launch_bounds__(128)
void attn_flash_kernel(const __bf16* __restrict__ q,
                       const __bf16* __restrict__ kk,
                       const __bf16* __restrict__ vT,
                       const int* __restrict__ causal_flag,
                       __bf16* __restrict__ attn) {
  const int lane = threadIdx.x & 31;
  const int wid  = threadIdx.x >> 5;
  const int bh = blockIdx.y;                         // 0..31
  const int qt = blockIdx.x * 4 + wid;               // 0..127
  const int q0 = qt * 16;
  const int causal = *causal_flag;

  const __bf16* qbase = q  + bh * SEQ * HDIM;        // [S,D] (true d order)
  const __bf16* kbase = kk + bh * SEQ * HDIM;        // [S,D]
  const __bf16* vbase = vT + bh * HDIM * SEQ;        // [D,S]

  // Q fragments (Q already * 1/sqrt(D), stored in true d order)
  v16bf qa0 = load_frag_a(qbase + q0 * HDIM + 0,  HDIM, lane);
  v16bf qa1 = load_frag_a(qbase + q0 * HDIM + 32, HDIM, lane);

  const int n = lane & 15;
  const int mrow0 = (lane < 16) ? 0 : 8;

  float m_i[8], l_i[8];
#pragma unroll
  for (int r = 0; r < 8; ++r) { m_i[r] = -1.0e30f; l_i[r] = 0.0f; }
  v8f o[4] = {};

  __shared__ __align__(32) __bf16 pshare[4][16 * 32];
  __bf16* pb = pshare[wid];

  const int nchunks = causal ? ((q0 + 47) >> 5) : (SEQ >> 5);

  // preload K fragments for chunk 0
  v16bf kf[4];
  kf[0] = load_frag_b(kbase + 0 * HDIM + 0,   HDIM, lane);
  kf[1] = load_frag_b(kbase + 0 * HDIM + 32,  HDIM, lane);
  kf[2] = load_frag_b(kbase + 16 * HDIM + 0,  HDIM, lane);
  kf[3] = load_frag_b(kbase + 16 * HDIM + 32, HDIM, lane);

#pragma unroll 1
  for (int c = 0; c < nchunks; ++c) {
    const int kc = c * 32;

    // V fragments: independent of P -> issue early, complete behind softmax.
    v16bf vb[4];
#pragma unroll
    for (int t = 0; t < 4; ++t) vb[t] = load_frag_b(vbase + (t * 16) * SEQ + kc, SEQ, lane);

    // --- S = Q @ K^T, two 16x16 tiles (keys kc+0..15, kc+16..31) ---
    v8f s0 = {}, s1 = {};
    s0 = WMMA_BF16(qa0, kf[0], s0);
    s0 = WMMA_BF16(qa1, kf[1], s0);
    s1 = WMMA_BF16(qa0, kf[2], s1);
    s1 = WMMA_BF16(qa1, kf[3], s1);

    // prefetch next chunk's K fragments in place, behind the softmax
    if (c + 1 < nchunks) {
      const int kn = kc + 32;
      kf[0] = load_frag_b(kbase + (kn + 0) * HDIM + 0,   HDIM, lane);
      kf[1] = load_frag_b(kbase + (kn + 0) * HDIM + 32,  HDIM, lane);
      kf[2] = load_frag_b(kbase + (kn + 16) * HDIM + 0,  HDIM, lane);
      kf[3] = load_frag_b(kbase + (kn + 16) * HDIM + 32, HDIM, lane);
    }

    // --- causal mask (uniform branch per wave) ---
    if (causal && (kc + 31 > q0)) {
#pragma unroll
      for (int r = 0; r < 8; ++r) {
        const int qrow = q0 + mrow0 + r;
        if (kc + n      > qrow) s0[r] = -1.0e30f;
        if (kc + 16 + n > qrow) s1[r] = -1.0e30f;
      }
    }

    // --- online softmax over the 32-key chunk ---
    float scale[8];
#pragma unroll
    for (int r = 0; r < 8; ++r) {
      float mx = fmaxf(s0[r], s1[r]);
      mx = fmaxf(mx, __shfl_xor(mx, 1, 32));
      mx = fmaxf(mx, __shfl_xor(mx, 2, 32));
      mx = fmaxf(mx, __shfl_xor(mx, 4, 32));
      mx = fmaxf(mx, __shfl_xor(mx, 8, 32));
      const float mnew = fmaxf(m_i[r], mx);
      const float p0 = __expf(s0[r] - mnew);
      const float p1 = __expf(s1[r] - mnew);
      scale[r] = __expf(m_i[r] - mnew);
      float rs = p0 + p1;
      rs += __shfl_xor(rs, 1, 32);
      rs += __shfl_xor(rs, 2, 32);
      rs += __shfl_xor(rs, 4, 32);
      rs += __shfl_xor(rs, 8, 32);
      l_i[r] = l_i[r] * scale[r] + rs;
      m_i[r] = mnew;
      s0[r] = p0;
      s1[r] = p1;
    }
#pragma unroll
    for (int t = 0; t < 4; ++t)
#pragma unroll
      for (int r = 0; r < 8; ++r) o[t][r] *= scale[r];

    // --- P (C-layout f32) -> A-fragment via LDS, stored in true K order ---
#pragma unroll
    for (int r = 0; r < 8; ++r) {
      const int m = mrow0 + r;
      pb[m * 32 + n]      = (__bf16)s0[r];
      pb[m * 32 + 16 + n] = (__bf16)s1[r];
    }
    asm volatile("s_wait_dscnt 0" ::: "memory");   // per-wave LDS RAW fence
    v16bf pa = *(const v16bf*)(pb + (lane & 15) * 32 + ((lane < 16) ? 0 : 16));

    // --- O += P @ V ---
#pragma unroll
    for (int t = 0; t < 4; ++t) o[t] = WMMA_BF16(pa, vb[t], o[t]);
  }

  // --- normalize and store to attn [B,S,E] (true col order: A-operand of proj) ---
  const int b_ = bh >> 4, h_ = bh & 15;
  float inv[8];
#pragma unroll
  for (int r = 0; r < 8; ++r) inv[r] = 1.0f / l_i[r];
#pragma unroll
  for (int t = 0; t < 4; ++t)
#pragma unroll
    for (int r = 0; r < 8; ++r) {
      const int m = mrow0 + r;
      const int col = h_ * HDIM + t * 16 + n;
      const int colswz = (col & ~31) | swz32(col & 31);
      attn[(b_ * SEQ + q0 + m) * EMB + col] = (__bf16)(o[t][r] * inv[r]);
    }
}

// ---------------------------------------------------------------------------
// Kernel 4: output projection, 32x64 tile per wave. f32 out + bias.
// ---------------------------------------------------------------------------
__global__ __launch_bounds__(128)
void attn_out_gemm_kernel(const __bf16* __restrict__ ab,
                          const __bf16* __restrict__ woutT,
                          const float* __restrict__ b_out,
                          float* __restrict__ out) {
  const int lane = threadIdx.x & 31;
  const int wid  = threadIdx.x >> 5;
  const int i0 = blockIdx.x * 32;
  const int j0 = (blockIdx.y * 4 + wid) * 64;

  v8f acc0[4] = {}, acc1[4] = {};
  gemm_core_32x64(ab + i0 * EMB, ab + (i0 + 16) * EMB, woutT + j0 * EMB,
                  lane, acc0, acc1);

  const int n = lane & 15;
  const int mrow0 = (lane < 16) ? 0 : 8;
#pragma unroll
  for (int t = 0; t < 4; ++t) {
    const int j = j0 + t * 16 + n;
    const float bv = b_out[j];
#pragma unroll
    for (int r = 0; r < 8; ++r) {
      out[(i0 + mrow0 + r) * EMB + j]      = acc0[t][r] + bv;
      out[(i0 + 16 + mrow0 + r) * EMB + j] = acc1[t][r] + bv;
    }
  }
}

// ---------------------------------------------------------------------------
// Launch. Workspace layout (bytes), total ~56 MiB:
//   xb @0, winT @16777216, woutT @23068672, q @25165824, k @33554432,
//   vT @41943040, attn @50331648
// ---------------------------------------------------------------------------
extern "C" void kernel_launch(void* const* d_in, const int* in_sizes, int n_in,
                              void* d_out, int out_size, void* d_ws, size_t ws_size,
                              hipStream_t stream) {
  const float* x    = (const float*)d_in[0];
  const float* Win  = (const float*)d_in[1];
  const float* bin  = (const float*)d_in[2];
  const float* Wout = (const float*)d_in[3];
  const float* bout = (const float*)d_in[4];
  const int*   cmsk = (const int*)d_in[5];

  char* ws = (char*)d_ws;
  __bf16* xb    = (__bf16*)(ws + 0);
  __bf16* winT  = (__bf16*)(ws + 16777216);
  __bf16* woutT = (__bf16*)(ws + 23068672);
  __bf16* qb    = (__bf16*)(ws + 25165824);
  __bf16* kb    = (__bf16*)(ws + 33554432);
  __bf16* vTb   = (__bf16*)(ws + 41943040);
  __bf16* attnb = (__bf16*)(ws + 50331648);

  // 1) pack/convert/transpose
  attn_pack_kernel<<<49152, 256, 0, stream>>>(x, Win, Wout, xb, winT, woutT);
  // 2) QKV projection: 128 row-tiles (32 rows) x 48 strips (12 blocks * 4 waves)
  attn_qkv_gemm_kernel<<<dim3(128, 12), 128, 0, stream>>>(xb, winT, bin, qb, kb, vTb);
  // 3) flash attention: 128 q-tiles (32 blocks * 4 waves) x 32 (b,h)
  attn_flash_kernel<<<dim3(32, 32), 128, 0, stream>>>(qb, kb, vTb, cmsk, attnb);
  // 4) output projection: 128 row-tiles x 16 strips (4 blocks * 4 waves)
  attn_out_gemm_kernel<<<dim3(128, 4), 128, 0, stream>>>(attnb, woutT, bout, (float*)d_out);
}
